// SRGNN_38517266710880
// MI455X (gfx1250) — matrix operand order repaired and checked by reading, not compiled, not run on hardware
//
#include <hip/hip_runtime.h>
#include <cstddef>
#include <cstdint>

// ---------------------------------------------------------------------------
// SR-GNN forward for MI455X (gfx1250, wave32, WMMA).
// All GEMMs run through v_wmma_f32_16x16x32_bf16 (bf16 A/B, f32 accumulate).
// ---------------------------------------------------------------------------

#define HH 128      // H
#define VV 100000   // V
#define BB 1024     // B
#define NN 50       // N
#define LL 50       // L

typedef __attribute__((ext_vector_type(16))) __bf16 v16bf;
typedef __attribute__((ext_vector_type(8)))  float  v8f;

#define DEV static __device__ __forceinline__

DEV __bf16 f2bf(float f) {
  // round-to-nearest-even f32 -> bf16 (bit twiddle; no reliance on hw cast)
  unsigned u = __builtin_bit_cast(unsigned, f);
  unsigned r = u + 0x7FFFu + ((u >> 16) & 1u);
  unsigned short h = (unsigned short)(r >> 16);
  return __builtin_bit_cast(__bf16, h);
}

DEV float sigmoidf(float x) { return 1.0f / (1.0f + __expf(-x)); }

// --- WMMA fragment loaders ---------------------------------------------------
// A fragment, 16x32 bf16, A row-major [M,K] (lda = K):
//   lane<16 : row = m0+lane,    K = {k0..k0+7} U {k0+16..k0+23}
//   lane>=16: row = m0+lane-16, K = {k0+8..k0+15} U {k0+24..k0+31}
DEV v16bf load_afrag(const __bf16* __restrict__ A, int lda, int m0, int k0, int lane) {
  const int half = lane >> 4;
  const __bf16* p = A + (size_t)(m0 + (lane & 15)) * lda + k0 + half * 8;
  v16bf r;
#pragma unroll
  for (int i = 0; i < 8; ++i) r[i] = p[i];
#pragma unroll
  for (int i = 0; i < 8; ++i) r[8 + i] = p[16 + i];
  return r;
}

// B fragment, 32x16 bf16, where B = W^T and W is row-major [Nout, K] (ldb = K):
//   lane<16 : col n = n0+lane,    K = {k0..k0+7} U {k0+16..k0+23}
//   lane>=16: col n = n0+lane-16, K = {k0+8..k0+15} U {k0+24..k0+31}
// OOB columns are CLAMPED to row N-1 (valid memory, garbage values); the
// store path masks those columns.  Branch-free -> EXEC untouched in K-loop.
DEV v16bf load_bfrag(const __bf16* __restrict__ W, int ldb, int n0, int k0,
                     int lane, int Nmax) {
  const int half = lane >> 4;
  int n = n0 + (lane & 15);
  n = (n < Nmax) ? n : (Nmax - 1);
  const __bf16* p = W + (size_t)n * ldb + k0 + half * 8;
  v16bf r;
#pragma unroll
  for (int i = 0; i < 8; ++i) r[i] = p[i];
#pragma unroll
  for (int i = 0; i < 8; ++i) r[8 + i] = p[16 + i];
  return r;
}

DEV v8f wmma_bf16(v16bf a, v16bf b, v8f c) {
  return __builtin_amdgcn_wmma_f32_16x16x32_bf16(false, a, false, b, (short)0, c,
                                                 false, false);
}

// --- f32 -> bf16 convert (grid-stride) --------------------------------------
__global__ void cvt_bf16_kernel(const float* __restrict__ src,
                                __bf16* __restrict__ dst, size_t n) {
  size_t i = (size_t)blockIdx.x * blockDim.x + threadIdx.x;
  size_t stride = (size_t)gridDim.x * blockDim.x;
  for (; i < n; i += stride) dst[i] = f2bf(src[i]);
}

// --- hidden = emb[items] (f32 + bf16 copies) --------------------------------
__global__ void gather_hidden_kernel(const int* __restrict__ items,
                                     const float* __restrict__ emb,
                                     float* __restrict__ hf,
                                     __bf16* __restrict__ hb) {
  const int row = blockIdx.x;     // b*N + n
  const int h = threadIdx.x;      // 0..H-1
  const int item = items[row];
  const float v = emb[(size_t)item * HH + h];
  hf[(size_t)row * HH + h] = v;
  hb[(size_t)row * HH + h] = f2bf(v);
}

// --- generic GEMM: C[M,N] = A_bf16[M,K] @ W_bf16[N,K]^T (+bias) -------------
// Wave computes a 16 x (16*TN) strip: A fragment loaded once per K-step and
// reused by TN WMMAs. 8 waves per block along N.
template <int TN, bool NT>
__global__ void __launch_bounds__(256) gemm_bf16_kernel(
    const __bf16* __restrict__ Amat, const __bf16* __restrict__ Wmat,
    const float* __restrict__ bias, float* __restrict__ C, int ldc, int M,
    int N, int K) {
  const int lane = threadIdx.x & 31;
  const int wv = threadIdx.x >> 5;
  const int n0 = (blockIdx.x * 8 + wv) * (16 * TN);
  const int m0 = blockIdx.y * 16;
  if (n0 >= N) return;  // wave-uniform: EXEC stays all-ones for WMMA
  v8f acc[TN];
#pragma unroll
  for (int t = 0; t < TN; ++t) acc[t] = {};
  for (int k0 = 0; k0 < K; k0 += 32) {
    v16bf a = load_afrag(Amat, K, m0, k0, lane);
#pragma unroll
    for (int t = 0; t < TN; ++t) {
      v16bf b = load_bfrag(Wmat, K, n0 + 16 * t, k0, lane, N);
      acc[t] = wmma_bf16(a, b, acc[t]);
    }
  }
  const int rbase = m0 + ((lane >> 4) << 3);
#pragma unroll
  for (int t = 0; t < TN; ++t) {
    const int col = n0 + 16 * t + (lane & 15);
    if (col < N) {
      const float bv = bias ? bias[col] : 0.0f;
#pragma unroll
      for (int r = 0; r < 8; ++r) {
        float v = acc[t][r] + bv;
        float* p = C + (size_t)(rbase + r) * ldc + col;
        if (NT)
          __builtin_nontemporal_store(v, p);  // 410MB stream: keep L2 for emb
        else
          *p = v;
      }
    }
  }
}

// --- adjacency aggregation: inputs = [A_in@hin + b_iah, A_out@hout + b_oah] -
// one block per (b,n) row; A row elements are wave-uniform scalar broadcasts.
__global__ void agg_kernel(const float* __restrict__ Amat,
                           const float* __restrict__ hin,
                           const float* __restrict__ hout,
                           const float* __restrict__ b_iah,
                           const float* __restrict__ b_oah,
                           __bf16* __restrict__ inputs_b) {
  const int row = blockIdx.x;  // b*N + n
  const int b = row / NN;
  const int h = threadIdx.x;   // 0..H-1
  const float* Arow = Amat + (size_t)row * (2 * NN);
  const float* pin = hin + (size_t)b * NN * HH + h;
  const float* pout = hout + (size_t)b * NN * HH + h;
  float accin = b_iah[h];
  float accout = b_oah[h];
#pragma unroll 5
  for (int j = 0; j < NN; ++j) {
    accin += Arow[j] * pin[(size_t)j * HH];
    accout += Arow[NN + j] * pout[(size_t)j * HH];
  }
  inputs_b[(size_t)row * (2 * HH) + h] = f2bf(accin);
  inputs_b[(size_t)row * (2 * HH) + HH + h] = f2bf(accout);
}

// --- fused GRU cell ----------------------------------------------------------
// Block handles 16 rows x all H output cols. Wave w owns cols [16w,16w+16) and
// computes the six gate tiles (gi_r/i/n via K=2H, gh_r/i/n via K=H) with WMMA,
// then applies the gates in-register. gi/gh never hit memory (~157MB saved).
__global__ void __launch_bounds__(256) gru_fused_kernel(
    const __bf16* __restrict__ inputs_b, const __bf16* __restrict__ hidden_b,
    const float* __restrict__ hidden_f, const __bf16* __restrict__ wih_b,
    const __bf16* __restrict__ whh_b, const float* __restrict__ b_ih,
    const float* __restrict__ b_hh, float* __restrict__ newh) {
  const int lane = threadIdx.x & 31;
  const int wv = threadIdx.x >> 5;  // 0..7
  const int m0 = blockIdx.x * 16;
  v8f gir = {}, gii = {}, gin_ = {}, ghr = {}, ghi = {}, ghn = {};
#pragma unroll
  for (int k0 = 0; k0 < 2 * HH; k0 += 32) {
    v16bf a = load_afrag(inputs_b, 2 * HH, m0, k0, lane);
    v16bf br = load_bfrag(wih_b, 2 * HH, wv * 16, k0, lane, 3 * HH);
    v16bf bi = load_bfrag(wih_b, 2 * HH, wv * 16 + HH, k0, lane, 3 * HH);
    v16bf bn = load_bfrag(wih_b, 2 * HH, wv * 16 + 2 * HH, k0, lane, 3 * HH);
    gir = wmma_bf16(a, br, gir);
    gii = wmma_bf16(a, bi, gii);
    gin_ = wmma_bf16(a, bn, gin_);
  }
#pragma unroll
  for (int k0 = 0; k0 < HH; k0 += 32) {
    v16bf a = load_afrag(hidden_b, HH, m0, k0, lane);
    v16bf br = load_bfrag(whh_b, HH, wv * 16, k0, lane, 3 * HH);
    v16bf bi = load_bfrag(whh_b, HH, wv * 16 + HH, k0, lane, 3 * HH);
    v16bf bn = load_bfrag(whh_b, HH, wv * 16 + 2 * HH, k0, lane, 3 * HH);
    ghr = wmma_bf16(a, br, ghr);
    ghi = wmma_bf16(a, bi, ghi);
    ghn = wmma_bf16(a, bn, ghn);
  }
  const int cg = wv * 16 + (lane & 15);           // output col 0..H-1
  const int rbase = m0 + ((lane >> 4) << 3);      // C/D layout: rows r / r+8
  const float bir = b_ih[cg], bii = b_ih[HH + cg], bin_ = b_ih[2 * HH + cg];
  const float bhr = b_hh[cg], bhi = b_hh[HH + cg], bhn = b_hh[2 * HH + cg];
#pragma unroll
  for (int r = 0; r < 8; ++r) {
    const int row = rbase + r;
    float ir = gir[r] + bir, ii = gii[r] + bii, in_ = gin_[r] + bin_;
    float hr = ghr[r] + bhr, hi = ghi[r] + bhi, hn = ghn[r] + bhn;
    float rg = sigmoidf(ir + hr);
    float ig = sigmoidf(ii + hi);
    float ng = tanhf(in_ + rg * hn);
    float hprev = hidden_f[(size_t)row * HH + cg];
    newh[(size_t)row * HH + cg] = ng + ig * (hprev - ng);
  }
}

// --- seq_hidden gather (bf16 for q2 GEMM) ------------------------------------
__global__ void seq_gather_kernel(const int* __restrict__ alias,
                                  const float* __restrict__ newh,
                                  __bf16* __restrict__ seq_b) {
  const int row = blockIdx.x;  // b*L + l
  const int b = row / LL;
  const int h = threadIdx.x;
  const int idx = alias[row];
  seq_b[(size_t)row * HH + h] =
      f2bf(newh[((size_t)b * NN + idx) * HH + h]);
}

// --- ht = seq_hidden[b, sum(mask)-1] -----------------------------------------
__global__ void ht_kernel(const int* __restrict__ alias,
                          const int* __restrict__ mask,
                          const float* __restrict__ newh,
                          __bf16* __restrict__ ht_b) {
  const int b = blockIdx.x;
  const int h = threadIdx.x;
  int s = 0;
  for (int l = 0; l < LL; ++l) s += mask[b * LL + l];
  const int last = s - 1;
  const int idx = alias[b * LL + last];
  ht_b[(size_t)b * HH + h] = f2bf(newh[((size_t)b * NN + idx) * HH + h]);
}

// --- attention pooling: alpha = sigmoid(q1+q2)@W_fc3^T + b_fc3;
//     a = sum_l alpha * seq_hidden * mask  (seq re-gathered from newh in f32)
__global__ void pool_kernel(const int* __restrict__ alias,
                            const int* __restrict__ mask,
                            const float* __restrict__ newh,
                            const float* __restrict__ q1,
                            const float* __restrict__ q2,
                            const float* __restrict__ W_fc3,
                            const float* __restrict__ b_fc3,
                            __bf16* __restrict__ a_b) {
  __shared__ float red[4];
  const int b = blockIdx.x;
  const int h = threadIdx.x;  // 0..H-1 (4 waves)
  const float w3 = W_fc3[h];
  const float q1v = q1[(size_t)b * HH + h];
  float acc = 0.0f;
  for (int l = 0; l < LL; ++l) {
    float s = sigmoidf(q1v + q2[((size_t)b * LL + l) * HH + h]) * w3;
#pragma unroll
    for (int off = 16; off >= 1; off >>= 1) s += __shfl_xor(s, off);
    if ((h & 31) == 0) red[h >> 5] = s;
    __syncthreads();
    const float alpha = red[0] + red[1] + red[2] + red[3] + b_fc3[0];
    __syncthreads();
    const float m = (float)mask[b * LL + l];
    const int idx = alias[b * LL + l];
    const float sh = newh[((size_t)b * NN + idx) * HH + h];
    acc += alpha * sh * m;
  }
  a_b[(size_t)b * HH + h] = f2bf(acc);
}

// ---------------------------------------------------------------------------
extern "C" void kernel_launch(void* const* d_in, const int* in_sizes, int n_in,
                              void* d_out, int out_size, void* d_ws,
                              size_t ws_size, hipStream_t stream) {
  (void)in_sizes; (void)n_in; (void)out_size; (void)ws_size;
  const int*   alias  = (const int*)d_in[0];
  const float* Amat   = (const float*)d_in[1];
  const int*   items  = (const int*)d_in[2];
  const int*   mask   = (const int*)d_in[3];
  const float* emb    = (const float*)d_in[4];
  const float* w_ih   = (const float*)d_in[5];
  const float* w_hh   = (const float*)d_in[6];
  const float* b_ih   = (const float*)d_in[7];
  const float* b_hh   = (const float*)d_in[8];
  const float* b_iah  = (const float*)d_in[9];
  const float* b_oah  = (const float*)d_in[10];
  const float* W_ein  = (const float*)d_in[11];
  const float* b_ein  = (const float*)d_in[12];
  const float* W_eout = (const float*)d_in[13];
  const float* b_eout = (const float*)d_in[14];
  const float* W_fc1  = (const float*)d_in[15];
  const float* b_fc1  = (const float*)d_in[16];
  const float* W_fc2  = (const float*)d_in[17];
  const float* b_fc2  = (const float*)d_in[18];
  const float* W_fc3  = (const float*)d_in[19];
  const float* b_fc3  = (const float*)d_in[20];
  float* scores = (float*)d_out;

  // workspace carve-out (~150 MB with aliasing)
  char* w = (char*)d_ws;
  auto alloc = [&](size_t bytes) {
    void* p = (void*)w;
    w += (bytes + 255) & ~(size_t)255;
    return p;
  };
  const size_t BNH = (size_t)BB * NN * HH;
  __bf16* emb_b    = (__bf16*)alloc((size_t)VV * HH * 2);
  float*  hidden_f = (float*)alloc(BNH * 4);
  __bf16* hidden_b = (__bf16*)alloc(BNH * 2);
  float*  hin      = (float*)alloc(BNH * 4);  // later reused as newh
  float*  hout     = (float*)alloc(BNH * 4);  // later reused as q2
  __bf16* inputs_b = (__bf16*)alloc(BNH * 2 * 2);
  __bf16* ht_b     = (__bf16*)alloc((size_t)BB * HH * 2);
  float*  q1       = (float*)alloc((size_t)BB * HH * 4);
  __bf16* a_b      = (__bf16*)alloc((size_t)BB * HH * 2);
  __bf16* wih_b    = (__bf16*)alloc((size_t)3 * HH * 2 * HH * 2);
  __bf16* whh_b    = (__bf16*)alloc((size_t)3 * HH * HH * 2);
  __bf16* Wein_b   = (__bf16*)alloc((size_t)HH * HH * 2);
  __bf16* Weout_b  = (__bf16*)alloc((size_t)HH * HH * 2);
  __bf16* Wfc1_b   = (__bf16*)alloc((size_t)HH * HH * 2);
  __bf16* Wfc2_b   = (__bf16*)alloc((size_t)HH * HH * 2);

  // buffer reuse after lifetimes end:
  float*  newh  = hin;              // hin dead after agg_kernel
  float*  q2    = hout;             // hout dead after agg_kernel
  __bf16* seq_b = hidden_b;         // hidden_b dead after gru_fused_kernel

  // 1) precision prep: f32 -> bf16 (emb is large; weights tiny)
  cvt_bf16_kernel<<<2048, 256, 0, stream>>>(emb, emb_b, (size_t)VV * HH);
  cvt_bf16_kernel<<<96, 256, 0, stream>>>(w_ih, wih_b, (size_t)3 * HH * 2 * HH);
  cvt_bf16_kernel<<<48, 256, 0, stream>>>(w_hh, whh_b, (size_t)3 * HH * HH);
  cvt_bf16_kernel<<<16, 256, 0, stream>>>(W_ein, Wein_b, (size_t)HH * HH);
  cvt_bf16_kernel<<<16, 256, 0, stream>>>(W_eout, Weout_b, (size_t)HH * HH);
  cvt_bf16_kernel<<<16, 256, 0, stream>>>(W_fc1, Wfc1_b, (size_t)HH * HH);
  cvt_bf16_kernel<<<16, 256, 0, stream>>>(W_fc2, Wfc2_b, (size_t)HH * HH);

  // 2) hidden = emb[items]
  gather_hidden_kernel<<<BB * NN, HH, 0, stream>>>(items, emb, hidden_f,
                                                   hidden_b);

  // 3) hin = hidden @ W_ein^T + b_ein ; hout = hidden @ W_eout^T + b_eout
  {
    dim3 g(1, (BB * NN) / 16);
    gemm_bf16_kernel<1, false><<<g, 256, 0, stream>>>(hidden_b, Wein_b, b_ein,
                                                      hin, HH, BB * NN, HH, HH);
    gemm_bf16_kernel<1, false><<<g, 256, 0, stream>>>(hidden_b, Weout_b, b_eout,
                                                      hout, HH, BB * NN, HH, HH);
  }

  // 4) adjacency aggregation -> inputs (bf16, biases folded in)
  agg_kernel<<<BB * NN, HH, 0, stream>>>(Amat, hin, hout, b_iah, b_oah,
                                         inputs_b);

  // 5) fused GRU: gi/gh WMMA + gates, write new hidden (f32)
  gru_fused_kernel<<<(BB * NN) / 16, 256, 0, stream>>>(
      inputs_b, hidden_b, hidden_f, wih_b, whh_b, b_ih, b_hh, newh);

  // 6) gathers for attention
  seq_gather_kernel<<<BB * LL, HH, 0, stream>>>(alias, newh, seq_b);
  ht_kernel<<<BB, HH, 0, stream>>>(alias, mask, newh, ht_b);

  // 7) q1 = ht @ W_fc1^T + b_fc1 ; q2 = seq_hidden @ W_fc2^T + b_fc2
  gemm_bf16_kernel<1, false><<<dim3(1, BB / 16), 256, 0, stream>>>(
      ht_b, Wfc1_b, b_fc1, q1, HH, BB, HH, HH);
  gemm_bf16_kernel<1, false><<<dim3(1, (BB * LL) / 16), 256, 0, stream>>>(
      seq_b, Wfc2_b, b_fc2, q2, HH, BB * LL, HH, HH);

  // 8) attention pooling -> a (bf16)
  pool_kernel<<<BB, HH, 0, stream>>>(alias, mask, newh, q1, q2, W_fc3, b_fc3,
                                     a_b);

  // 9) scores = a @ emb[1:]^T  — 16x64 strip per wave (TN=4): A fragment
  //    reused by 4 WMMAs per K-step; NT stores keep emb resident in L2.
  {
    const int Nout = VV - 1;
    const int nstrips = (Nout + 63) / 64;       // 64 cols per wave
    dim3 g((nstrips + 7) / 8, BB / 16);
    gemm_bf16_kernel<4, true><<<g, 256, 0, stream>>>(a_b, emb_b + HH, nullptr,
                                                     scores, Nout, BB, Nout, HH);
  }
}
